// CSRGridBuilder_21766894256160
// MI455X (gfx1250) — compile-verified, matrix operand order checked by reading, not run on hardware
//
#include <hip/hip_runtime.h>
#include <hip/hip_bf16.h>

// ---------------------------------------------------------------------------
// CSR grid builder for MI455X (gfx1250, wave32).
//
// Pipeline (deterministic, graph-capturable, stream-only):
//   1. prep:      per-sphere R(quat), AABB, cov_inv, norm_factor   (VALU)
//   2. quantiles: 7 exact order statistics via radix sort on
//                 monotone float->u32 keys                          (sort)
//   3. pairgen:   16.7M packed (morton<<32|id) records              (VMEM)
//   4. sort:      4-pass stable LSD radix. Tiles are staged into
//                 LDS by the Tensor Data Mover (tensor_load_to_lds,
//                 TENSORcnt) so the DMA overlaps the wave32
//                 ballot-match ranking; no global atomics in the
//                 data path -> bit-deterministic.                   (sort)
//   5. emit:      sm/ss/l1_offsets/num_unique (+global_prefetch)    (VMEM)
//   6. total_pairs: WMMA ones-matmul reduction on the matrix pipe
//                 (counts<=64 exact in f16; partials are integer
//                 multiples of 16 below 2^28 -> exact f32)
// ---------------------------------------------------------------------------

typedef __attribute__((ext_vector_type(16))) _Float16 v16h;
typedef __attribute__((ext_vector_type(8)))  float    v8f;

// TDM descriptor group types (clang-23 / therock 6-arg builtin signature:
//  (uint32x4 g0, int32x8 g1, int32x4, int32x4, int32x8, i32 cpol)).
typedef unsigned int t4u __attribute__((ext_vector_type(4)));
typedef int          t8i __attribute__((ext_vector_type(8)));
typedef int          t4i __attribute__((ext_vector_type(4)));

#define TILE      2048          // elements per radix tile (256 thr x 8 items)
#define SENT_KEY  (1u << 30)

// ------------------------------ helpers ------------------------------------

__device__ __forceinline__ unsigned int expand10(unsigned int x) {
  x &= 0x3FFu;
  x = (x | (x << 16)) & 0x030000FFu;
  x = (x | (x << 8))  & 0x0300F00Fu;
  x = (x | (x << 4))  & 0x030C30C3u;
  x = (x | (x << 2))  & 0x09249249u;
  return x;
}

__device__ __forceinline__ unsigned int mapf(float x) {
  unsigned int u = __float_as_uint(x);
  return (u & 0x80000000u) ? ~u : (u | 0x80000000u);
}
__device__ __forceinline__ float invmapf(unsigned int m) {
  unsigned int u = (m & 0x80000000u) ? (m ^ 0x80000000u) : ~m;
  return __uint_as_float(u);
}

// TDM: DMA one contiguous tile of n_elems u64 records into LDS.
// D# per CDNA5 ISA:
//   group0: count=1 | lds_addr | global_addr[56:0] | type=2
//   group1: data_size=3 (8B), tensor_dim0=tensor_elems, tensor_dim1=1,
//           tile_dim0=n_elems, tile_dim1=1, tensor_dim0_stride=tensor_elems
//   group2/3: zero (<=2D tensor)
// Generic->LDS address: ISA 10.2 maps LDS aperture addresses by truncation
// to the low 32 bits, which is the LDS byte offset.
__device__ __forceinline__ void tdm_load_tile(const unsigned long long* gsrc,
                                              void* lds_dst,
                                              unsigned int n_elems,
                                              unsigned int tensor_elems) {
  unsigned int lds_off = (unsigned int)(unsigned long long)lds_dst;
  unsigned long long ga = (unsigned long long)gsrc;
  t4u g0;
  g0.x = 1u;                                                  // count=1
  g0.y = lds_off;                                             // lds_addr
  g0.z = (unsigned int)ga;                                    // global_addr lo
  g0.w = (unsigned int)((ga >> 32) & 0x01FFFFFFu) | (2u << 30); // hi | type=2
  t8i g1;
  g1[0] = (int)(3u << 16);                                    // data_size=8B
  g1[1] = (int)((tensor_elems & 0xFFFFu) << 16);              // td0[15:0]
  g1[2] = (int)((tensor_elems >> 16) | (1u << 16));           // td0[31:16]|td1=1
  g1[3] = (int)(n_elems << 16);                               // tile_dim0
  g1[4] = 1;                                                  // tile_dim1=1
  g1[5] = (int)tensor_elems;                                  // stride0 lo
  g1[6] = 0;                                                  // stride0 hi
  g1[7] = 0;
  t4i z4 = { 0, 0, 0, 0 };
  t8i z8 = { 0, 0, 0, 0, 0, 0, 0, 0 };
  __builtin_amdgcn_tensor_load_to_lds(g0, g1, z4, z4, z8, 0);
}

// ------------------------- 1. per-sphere prep -------------------------------

__global__ void csr_prep(const float* __restrict__ pos, const float* __restrict__ scl,
                         const float* __restrict__ qua,
                         float* __restrict__ minc, float* __restrict__ maxc,
                         float* __restrict__ rad0,
                         float* __restrict__ o_cov, float* __restrict__ o_nf, int M) {
  int s = blockIdx.x * blockDim.x + threadIdx.x;
  if (s >= M) return;
  float w = qua[s*4+0], x = qua[s*4+1], y = qua[s*4+2], z = qua[s*4+3];
  float nrm = sqrtf(w*w + x*x + y*y + z*z) + 1e-7f;
  w /= nrm; x /= nrm; y /= nrm; z /= nrm;
  float R[3][3];
  R[0][0]=1.f-2.f*(y*y+z*z); R[0][1]=2.f*(x*y-w*z); R[0][2]=2.f*(x*z+w*y);
  R[1][0]=2.f*(x*y+w*z); R[1][1]=1.f-2.f*(x*x+z*z); R[1][2]=2.f*(y*z-w*x);
  R[2][0]=2.f*(x*z-w*y); R[2][1]=2.f*(y*z+w*x); R[2][2]=1.f-2.f*(x*x+y*y);
  float s0 = scl[s*3+0], s1 = scl[s*3+1], s2 = scl[s*3+2];
  float sq[3] = { s0*s0, s1*s1, s2*s2 };
  const float K = 2.7954835f;                 // sqrt(chi2.ppf(0.95, 3))
  for (int i = 0; i < 3; ++i) {
    float acc = R[i][0]*R[i][0]*sq[0] + R[i][1]*R[i][1]*sq[1] + R[i][2]*R[i][2]*sq[2];
    float r = K * sqrtf(acc + 1e-12f);
    float p = pos[s*3+i];
    minc[s*3+i] = p - r;
    maxc[s*3+i] = p + r;
    if (i == 0) rad0[s] = r;
  }
  float iv[3] = { 1.f/(sq[0]+1e-12f), 1.f/(sq[1]+1e-12f), 1.f/(sq[2]+1e-12f) };
  for (int i = 0; i < 3; ++i)
    for (int k = 0; k < 3; ++k)
      o_cov[(size_t)s*9 + i*3 + k] =
          R[i][0]*iv[0]*R[k][0] + R[i][1]*iv[1]*R[k][1] + R[i][2]*iv[2]*R[k][2];
  o_nf[s] = 1.f / (15.749609945722419f * s0*s1*s2 + 1e-7f);   // (2*pi)^1.5
}

// --------------------------- radix sort core --------------------------------

__global__ void radix_hist(const unsigned long long* __restrict__ in,
                           unsigned int* __restrict__ hist,
                           int nb, int shift, int n) {
  __shared__ unsigned long long tile[TILE];
  __shared__ unsigned int c[256];
  int t = threadIdx.x, b = blockIdx.x;
  c[t] = 0;
  if (t < 32) {                 // wave 0 issues the tile DMA (TDM ignores EXEC)
    tdm_load_tile(in + (size_t)b * TILE, tile, TILE, (unsigned int)n);
    __builtin_amdgcn_s_wait_tensorcnt(0);
  }
  __syncthreads();
  for (int j = 0; j < 8; ++j) {
    unsigned int d = (unsigned int)((tile[j*256 + t] >> shift) & 0xFFull);
    atomicAdd(&c[d], 1u);
  }
  __syncthreads();
  hist[(size_t)t * nb + b] = c[t];               // digit-major layout
}

__global__ void scan_a(unsigned int* __restrict__ data,
                       unsigned int* __restrict__ sums) {
  __shared__ unsigned int lds[256];
  int t = threadIdx.x, b = blockIdx.x;
  size_t base = (size_t)b*1024 + t*4;
  unsigned int x0 = data[base], x1 = data[base+1], x2 = data[base+2], x3 = data[base+3];
  unsigned int tot = x0 + x1 + x2 + x3;
  lds[t] = tot; __syncthreads();
  for (int off = 1; off < 256; off <<= 1) {
    unsigned int v = (t >= off) ? lds[t-off] : 0u;
    __syncthreads();
    lds[t] += v;
    __syncthreads();
  }
  unsigned int ex = (t == 0) ? 0u : lds[t-1];
  data[base]   = ex;
  data[base+1] = ex + x0;
  data[base+2] = ex + x0 + x1;
  data[base+3] = ex + x0 + x1 + x2;
  if (t == 255) sums[b] = lds[255];
}

__global__ void scan_b(unsigned int* __restrict__ sums, int S) {
  __shared__ unsigned int lds[256];
  int t = threadIdx.x;
  int C = (S + 255) / 256;
  int base = t * C;
  unsigned int tot = 0;
  for (int i = 0; i < C; ++i) { int k = base + i; if (k < S) tot += sums[k]; }
  lds[t] = tot; __syncthreads();
  for (int off = 1; off < 256; off <<= 1) {
    unsigned int v = (t >= off) ? lds[t-off] : 0u;
    __syncthreads();
    lds[t] += v;
    __syncthreads();
  }
  unsigned int run = (t == 0) ? 0u : lds[t-1];
  for (int i = 0; i < C; ++i) {
    int k = base + i;
    if (k < S) { unsigned int x = sums[k]; sums[k] = run; run += x; }
  }
}

__global__ void scan_c(unsigned int* __restrict__ data,
                       const unsigned int* __restrict__ sums) {
  int b = blockIdx.x;
  unsigned int add = sums[b];
  size_t base = (size_t)b*1024 + threadIdx.x*4;
  data[base]   += add;
  data[base+1] += add;
  data[base+2] += add;
  data[base+3] += add;
}

// Stable scatter: TDM stages the tile into LDS; within-wave rank via 8-round
// ballot matching (wave32), cross-wave rank via LDS per-round counts,
// cross-round via running counts.
__global__ void radix_scatter(const unsigned long long* __restrict__ in,
                              unsigned long long* __restrict__ out,
                              const unsigned int* __restrict__ hist,
                              int nb, int shift, int n) {
  __shared__ unsigned long long tile[TILE];
  __shared__ unsigned int gbase[256];
  __shared__ unsigned int runc[256];
  __shared__ unsigned int c8[8][256];
  int t = threadIdx.x, b = blockIdx.x;
  int lane = t & 31, wave = t >> 5;
  gbase[t] = hist[(size_t)t * nb + b];
  runc[t] = 0;
  if (t < 32) {
    tdm_load_tile(in + (size_t)b * TILE, tile, TILE, (unsigned int)n);
    __builtin_amdgcn_s_wait_tensorcnt(0);
  }
  __syncthreads();
  for (int j = 0; j < 8; ++j) {
    for (int w = 0; w < 8; ++w) c8[w][t] = 0;
    __syncthreads();
    unsigned long long v = tile[j*256 + t];
    unsigned int d = (unsigned int)((v >> shift) & 0xFFull);
    unsigned int mask = 0xFFFFFFFFu;
    for (int bit = 0; bit < 8; ++bit) {
      unsigned int bm = (unsigned int)__ballot(((d >> bit) & 1u) != 0u);
      mask &= ((d >> bit) & 1u) ? bm : ~bm;
    }
    unsigned int laneRank = __popc(mask & ((1u << lane) - 1u));
    if (laneRank == 0u) c8[wave][d] = __popc(mask);
    __syncthreads();
    unsigned int waveBase = 0;
    for (int w = 0; w < wave; ++w) waveBase += c8[w][d];
    unsigned int rank = gbase[d] + runc[d] + waveBase + laneRank;
    out[rank] = v;
    __syncthreads();
    unsigned int tot = 0;
    for (int w = 0; w < 8; ++w) tot += c8[w][t];
    runc[t] += tot;
    __syncthreads();
  }
}

// --------------------------- quantile stage ---------------------------------

__global__ void pack_float(const float* __restrict__ src, int stride, int off,
                           unsigned long long* __restrict__ dst, int n) {
  int i = blockIdx.x * blockDim.x + threadIdx.x;
  if (i >= n) return;
  unsigned int m = mapf(src[(size_t)i*stride + off]);
  dst[i] = ((unsigned long long)m << 32) | (unsigned int)i;
}

__global__ void select_quantile(const unsigned long long* __restrict__ buf,
                                int n, double q, float* __restrict__ dst) {
  double pos = q * (double)(n - 1);
  int f = (int)pos;
  double frac = pos - (double)f;
  int f1 = (f + 1 < n) ? f + 1 : n - 1;
  float a = invmapf((unsigned int)(buf[f]  >> 32));
  float b = invmapf((unsigned int)(buf[f1] >> 32));
  *dst = (float)((double)a + frac * ((double)b - (double)a));
}

__global__ void finalize_bounds(float* __restrict__ scal,
                                float* __restrict__ o_gmin, float* __restrict__ o_vox) {
  for (int a = 0; a < 3; ++a) {
    float gmin = scal[a], gmax = scal[3 + a];
    float gm = gmin - 0.1f * (gmax - gmin);
    scal[8 + a] = gm;
    o_gmin[a] = gm;
  }
  float vox = scal[6] * 3.0f;
  scal[11] = vox;
  o_vox[0] = vox;
}

// ------------------------------ pair gen ------------------------------------

__global__ void pairgen(const float* __restrict__ minc, const float* __restrict__ maxc,
                        const float* __restrict__ scal,
                        unsigned long long* __restrict__ pairs,
                        float* __restrict__ o_ovr, _Float16* __restrict__ cnt, int M) {
  int tid = blockIdx.x * blockDim.x + threadIdx.x;
  int s = tid >> 6, o = tid & 63;
  if (s >= M) return;
  float vox = scal[11];
  int g0[3], g1[3];
  for (int a = 0; a < 3; ++a) {
    float gm = scal[8 + a];
    int lo = (int)floorf((minc[s*3+a] - gm) / vox);
    int hi = (int)floorf((maxc[s*3+a] - gm) / vox);
    g0[a] = min(1023, max(0, lo));
    g1[a] = min(1023, max(0, hi));
  }
  int e0 = g1[0]-g0[0]+1, e1 = g1[1]-g0[1]+1, e2 = g1[2]-g0[2]+1;
  int nv = e0 * e1 * e2;
  bool ovr = (nv > 64) || (e0 > 4) || (e1 > 4) || (e2 > 4);
  if (o == 0) {
    o_ovr[s] = ovr ? 1.0f : 0.0f;
    cnt[s] = (_Float16)(float)(ovr ? 0 : nv);
  }
  int ix = o >> 4, iy = (o >> 2) & 3, iz = o & 3;
  bool valid = !ovr && (ix < e0) && (iy < e1) && (iz < e2);
  unsigned long long v;
  if (valid) {
    unsigned int cx = (unsigned int)min(g0[0] + ix, 1023);
    unsigned int cy = (unsigned int)min(g0[1] + iy, 1023);
    unsigned int cz = (unsigned int)min(g0[2] + iz, 1023);
    unsigned int mort = (expand10(cx) << 2) | (expand10(cy) << 1) | expand10(cz);
    v = ((unsigned long long)mort << 32) | (unsigned int)s;
  } else {
    v = ((unsigned long long)SENT_KEY << 32) | 0xFFFFFFFFull;
  }
  pairs[(size_t)s * 64 + o] = v;
}

// ---------------- WMMA ones-matmul reduction for total_pairs ----------------
// D = A x 1: each row of D holds the exact row-sum of A. Counts<=64 are exact
// in f16, accumulators stay integer multiples of 16 below 2^28 -> exact f32,
// so the float atomicAdd is order-independent and deterministic.

__global__ void wmma_total_pairs(const _Float16* __restrict__ cnt,
                                 float* __restrict__ tp, int M) {
  int gwave  = (int)((blockIdx.x * blockDim.x + threadIdx.x) >> 5);
  int nwaves = (int)((gridDim.x * blockDim.x) >> 5);
  int lane = threadIdx.x & 31;
  v16h ones;
  for (int i = 0; i < 16; ++i) ones[i] = (_Float16)1.0f;
  v8f c = {};
  for (int chunk = gwave; chunk * 512 < M; chunk += nwaves) {
    const v16h* p = (const v16h*)(cnt + (size_t)chunk * 512);
    v16h a = p[lane];
    c = __builtin_amdgcn_wmma_f32_16x16x32_f16(
        false, a, false, ones, (short)0, c, false, false);
  }
  float s = c[0]+c[1]+c[2]+c[3]+c[4]+c[5]+c[6]+c[7];
  for (int off = 16; off > 0; off >>= 1) s += __shfl_xor(s, off, 32);
  s *= (1.0f / 16.0f);                  // each row-sum replicated over 16 cols
  if (lane == 0) atomicAdd(tp, s);
}

// --------------------------- emit / finalize --------------------------------

__global__ void csr_init(float* __restrict__ o_l1, float* __restrict__ scal) {
  int i = blockIdx.x * blockDim.x + threadIdx.x;
  if (i < 32768) o_l1[i] = -1.0f;
  if (i == 0) {
    scal[12] = 0.0f;                    // total_pairs accumulator (float)
    ((int*)(scal + 16))[0] = 0;         // num_unique counter (int)
  }
}

__global__ void csr_emit(const unsigned long long* __restrict__ pairs,
                         float* __restrict__ o_sm, float* __restrict__ o_ss,
                         float* __restrict__ o_l1, int* __restrict__ nu, int n) {
  int i = blockIdx.x * blockDim.x + threadIdx.x;
  if (i >= n) return;
  __builtin_prefetch(&pairs[i + 4096], 0, 0);   // global_prefetch_b8
  unsigned long long v = pairs[i];
  unsigned int key = (unsigned int)(v >> 32);
  o_sm[i] = (float)key;
  o_ss[i] = (float)(int)(unsigned int)(v & 0xFFFFFFFFull);
  bool valid = key < SENT_KEY;
  unsigned int prev = (i > 0) ? (unsigned int)(pairs[i-1] >> 32) : 0xFFFFFFFFu;
  if (valid && (i == 0 || key != prev)) atomicAdd(nu, 1);
  unsigned int cell = key >> 15;
  if (valid && (i == 0 || (prev >> 15) != cell)) o_l1[cell] = (float)i;
}

__global__ void finalize_scalars(const float* __restrict__ scal,
                                 float* __restrict__ o_tp, float* __restrict__ o_nu) {
  o_tp[0] = scal[12];
  o_nu[0] = (float)(((const int*)(scal + 16))[0]);
}

// ------------------------------ host driver ---------------------------------

static void run_radix_sort(unsigned long long*& A, unsigned long long*& B,
                           unsigned int* hist, unsigned int* bsums,
                           int n, hipStream_t stream) {
  int nb = n / TILE;                    // both n's are multiples of TILE
  int L  = 256 * nb;
  int sA = L / 1024;
  for (int pass = 0; pass < 4; ++pass) {
    int shift = 32 + pass * 8;
    radix_hist   <<<nb, 256, 0, stream>>>(A, hist, nb, shift, n);
    scan_a       <<<sA, 256, 0, stream>>>(hist, bsums);
    scan_b       <<<1,  256, 0, stream>>>(bsums, sA);
    scan_c       <<<sA, 256, 0, stream>>>(hist, bsums);
    radix_scatter<<<nb, 256, 0, stream>>>(A, B, hist, nb, shift, n);
    unsigned long long* t = A; A = B; B = t;
  }
}

extern "C" void kernel_launch(void* const* d_in, const int* in_sizes, int n_in,
                              void* d_out, int out_size, void* d_ws, size_t ws_size,
                              hipStream_t stream) {
  const float* pos = (const float*)d_in[0];
  const float* scl = (const float*)d_in[1];
  const float* qua = (const float*)d_in[2];
  const int M  = in_sizes[0] / 3;       // 262144
  const int NP = M * 64;                // 16777216

  // ---- output layout (reference return-tuple order, flattened as float) ----
  float* out    = (float*)d_out;
  float* o_sm   = out;
  float* o_ss   = o_sm  + (size_t)NP;
  float* o_ovr  = o_ss  + (size_t)NP;
  float* o_l1   = o_ovr + M;
  float* o_tp   = o_l1  + 32768;
  float* o_nu   = o_tp  + 1;
  float* o_gmin = o_nu  + 1;
  float* o_vox  = o_gmin + 3;
  float* o_cov  = o_vox + 1;
  float* o_nf   = o_cov + (size_t)M * 9;

  // ---- workspace layout (256B aligned slices) ----
  char* w = (char*)d_ws;
  size_t off = 0;
  auto au = [](size_t x) { return (x + 255) & ~(size_t)255; };
  unsigned long long* pA = (unsigned long long*)(w + off); off += au((size_t)NP * 8);
  unsigned long long* pB = (unsigned long long*)(w + off); off += au((size_t)NP * 8);
  unsigned int* hist  = (unsigned int*)(w + off); off += au((size_t)256 * (NP / TILE) * 4);
  unsigned int* bsums = (unsigned int*)(w + off); off += au((size_t)8192 * 4);
  float* minc = (float*)(w + off); off += au((size_t)M * 3 * 4);
  float* maxc = (float*)(w + off); off += au((size_t)M * 3 * 4);
  float* rad0 = (float*)(w + off); off += au((size_t)M * 4);
  _Float16* cnt = (_Float16*)(w + off); off += au((size_t)M * 2);
  float* scal = (float*)(w + off); off += au(256);
  (void)ws_size; (void)n_in; (void)out_size;

  // 0) reset counters / l1 table
  csr_init<<<128, 256, 0, stream>>>(o_l1, scal);

  // 1) per-sphere geometry + direct outputs (cov_inv, norm_factor)
  csr_prep<<<M / 256, 256, 0, stream>>>(pos, scl, qua, minc, maxc, rad0,
                                        o_cov, o_nf, M);

  // 2) exact quantiles via small radix sorts (float -> monotone u32 keys)
  const float* srcs[7]  = { minc, minc, minc, maxc, maxc, maxc, rad0 };
  const int   strides[7] = { 3, 3, 3, 3, 3, 3, 1 };
  const int   offs[7]    = { 0, 1, 2, 0, 1, 2, 0 };
  const double qs[7]     = { 0.01, 0.01, 0.01, 0.99, 0.99, 0.99, 0.5 };
  for (int s = 0; s < 7; ++s) {
    pack_float<<<M / 256, 256, 0, stream>>>(srcs[s], strides[s], offs[s], pA, M);
    unsigned long long* A = pA; unsigned long long* B = pB;
    run_radix_sort(A, B, hist, bsums, M, stream);
    select_quantile<<<1, 1, 0, stream>>>(A, M, qs[s], scal + s);
  }
  finalize_bounds<<<1, 1, 0, stream>>>(scal, o_gmin, o_vox);

  // 3) sphere-voxel pair generation (+ oversized flags + per-sphere counts)
  pairgen<<<NP / 256, 256, 0, stream>>>(minc, maxc, scal, pA, o_ovr, cnt, M);

  // 4) total_pairs on the matrix pipe (ones-matmul reduction, exact)
  wmma_total_pairs<<<64, 256, 0, stream>>>(cnt, scal + 12, M);

  // 5) sort 16.7M records by morton (stable -> matches jnp.argsort order)
  unsigned long long* A = pA; unsigned long long* B = pB;
  run_radix_sort(A, B, hist, bsums, NP, stream);

  // 6) sm/ss outputs, L1 boundaries, unique count
  csr_emit<<<NP / 256, 256, 0, stream>>>(A, o_sm, o_ss, o_l1,
                                         (int*)(scal + 16), NP);

  // 7) scalar outputs
  finalize_scalars<<<1, 1, 0, stream>>>(scal, o_tp, o_nu);
}
